// MultiheadSelfAttention_3702261809281
// MI455X (gfx1250) — compile-verified, matrix operand order checked by reading, not run on hardware
//
#include <hip/hip_runtime.h>

typedef __bf16 bf16;
typedef __attribute__((ext_vector_type(16))) __bf16 v16bf;
typedef __attribute__((ext_vector_type(8)))  float  v8f;
typedef __attribute__((ext_vector_type(4)))  float  v4f;
typedef __attribute__((ext_vector_type(4)))  unsigned int v4u;
typedef __attribute__((ext_vector_type(8)))  int    v8i;
typedef __attribute__((ext_vector_type(4)))  int    v4i;

#define D_MODEL 1024
#define N_HEADS 16
#define HEAD_DIM 64
#define SEQ 2048
#define BATCH 2

// ---------------------------------------------------------------------------
// TDM: 2D tile load Global -> LDS (bf16 elements, data_size = 2 bytes).
// D# per CDNA5 ISA. Issued per-wave (EXEC ignored),
// tracked with TENSORcnt.  6-arg builtin form (clang-23 / therock-10.0).
// ---------------------------------------------------------------------------
__device__ __forceinline__ void tdm_load_2d(
    const void* lds_dst, const void* gsrc,
    unsigned tensor_d0, unsigned tensor_d1,
    unsigned tile_d0, unsigned tile_d1,
    unsigned long long stride0_elems) {
  const unsigned long long ga = (unsigned long long)(size_t)gsrc;
  const unsigned lds = (unsigned)(size_t)lds_dst;  // low 32 bits = LDS offset
  v4u g0;
  g0.x = 1u;                                        // count=1, user mode
  g0.y = lds;                                       // lds_addr (bytes)
  g0.z = (unsigned)ga;                              // global_addr[31:0]
  g0.w = (unsigned)((ga >> 32) & 0x01FFFFFFu)       // global_addr[56:32]
         | (2u << 30);                              // type = 2 ("image")
  v8i g1;
  g1[0] = (int)(1u << 16);                          // wg_mask=0, data_size=1 (2B)
  g1[1] = (int)((tensor_d0 & 0xFFFFu) << 16);       // tensor_dim0[15:0] @ 63:48
  g1[2] = (int)((tensor_d0 >> 16) | ((tensor_d1 & 0xFFFFu) << 16));
  g1[3] = (int)((tensor_d1 >> 16) | ((tile_d0 & 0xFFFFu) << 16));
  g1[4] = (int)(tile_d1 & 0xFFFFu);                 // tile_dim1; tile_dim2=0
  g1[5] = (int)(unsigned)(stride0_elems & 0xFFFFFFFFull);   // dim0_stride lo
  g1[6] = (int)(unsigned)((stride0_elems >> 32) & 0xFFFFull); // stride hi; dim1_stride=0
  g1[7] = 0;
  const v4i zero4 = {0, 0, 0, 0};                   // groups 2/3 unused (2D)
  const v8i zero8 = {0, 0, 0, 0, 0, 0, 0, 0};
  __builtin_amdgcn_tensor_load_to_lds(g0, g1, zero4, zero4, zero8, 0);
}

// ---------------------------------------------------------------------------
// 16x32 bf16 fragment load from LDS (A m-major / B n-major identical form).
// Per ISA layout: lanes 0-15: K=0..7 (v0-3), K=16..23 (v4-7);
// lanes 16-31: K=8..15, K=24..31.  Two ds_load_b128 per lane.
// ---------------------------------------------------------------------------
__device__ __forceinline__ v16bf load_frag_k32(const bf16* base, int stride) {
  const int lane = threadIdx.x & 31;
  const int idx  = lane & 15;
  const int half = lane >> 4;
  const bf16* r = base + idx * stride + half * 8;
  union { v16bf v; v4u q[2]; } f;
  f.q[0] = *(const v4u*)(r);
  f.q[1] = *(const v4u*)(r + 16);
  return f.v;
}

// ---------------------------------------------------------------------------
// fp32 -> bf16 bulk convert (8 elements/thread, 16B stores)
// ---------------------------------------------------------------------------
__global__ __launch_bounds__(256) void cvt_f32_bf16(
    const float* __restrict__ s, bf16* __restrict__ d) {
  const size_t i = ((size_t)blockIdx.x * 256 + threadIdx.x) * 8;
  const v4f f0 = *(const v4f*)(s + i);
  const v4f f1 = *(const v4f*)(s + i + 4);
  union { bf16 b[8]; v4u u; } o;
  o.b[0] = (bf16)f0.x; o.b[1] = (bf16)f0.y; o.b[2] = (bf16)f0.z; o.b[3] = (bf16)f0.w;
  o.b[4] = (bf16)f1.x; o.b[5] = (bf16)f1.y; o.b[6] = (bf16)f1.z; o.b[7] = (bf16)f1.w;
  *(v4u*)(d + i) = o.u;
}

// ---------------------------------------------------------------------------
// C[M,N] = A[M,K] * B[N,K]^T, all bf16 in, f32 accumulate, bf16 or f32 out.
// 256 threads (8 waves), block tile 128x128, wave tile 32x64, K-step 32.
// Tiles fetched by TDM with double-buffered LDS (TENSORcnt pipelining).
// ---------------------------------------------------------------------------
template <bool OUT_BF16>
__global__ __launch_bounds__(256) void gemm_bf16_tdm(
    const bf16* __restrict__ A, const bf16* __restrict__ B,
    void* __restrict__ Cout, int M, int N, int K) {
  __shared__ __align__(16) bf16 sA[2][128 * 32];  // [m][k]
  __shared__ __align__(16) bf16 sB[2][128 * 32];  // [n][k]
  const int bm = blockIdx.x * 128;
  const int bn = blockIdx.y * 128;
  const int tid  = threadIdx.x;
  const int wave = tid >> 5;
  const int lane = tid & 31;
  const int wm = (wave & 3) * 32;   // wave row block (2 x 16)
  const int wn = (wave >> 2) * 64;  // wave col block (4 x 16)

  v8f acc[2][4] = {};

  if (wave == 0) {
    tdm_load_2d(sA[0], A + (size_t)bm * K, K, M, 32, 128, (unsigned)K);
    tdm_load_2d(sB[0], B + (size_t)bn * K, K, N, 32, 128, (unsigned)K);
  }

  int ib = 0;
  for (int k0 = 0; k0 < K; k0 += 32, ib ^= 1) {
    const bool more = (k0 + 32) < K;
    if (wave == 0) {
      if (more) {
        tdm_load_2d(sA[ib ^ 1], A + (size_t)bm * K + k0 + 32, K, M, 32, 128, (unsigned)K);
        tdm_load_2d(sB[ib ^ 1], B + (size_t)bn * K + k0 + 32, K, N, 32, 128, (unsigned)K);
        __builtin_amdgcn_s_wait_tensorcnt(2);  // the 2 just-issued may remain
      } else {
        __builtin_amdgcn_s_wait_tensorcnt(0);
      }
    }
    __syncthreads();  // current buffer ready for all waves

    const bf16* a_base = sA[ib];
    const bf16* b_base = sB[ib];
    const v16bf a0 = load_frag_k32(a_base + wm * 32, 32);
    const v16bf a1 = load_frag_k32(a_base + (wm + 16) * 32, 32);
#pragma unroll
    for (int t = 0; t < 4; ++t) {
      const v16bf b = load_frag_k32(b_base + (wn + t * 16) * 32, 32);
      acc[0][t] = __builtin_amdgcn_wmma_f32_16x16x32_bf16(
          false, a0, false, b, (short)0, acc[0][t], false, false);
      acc[1][t] = __builtin_amdgcn_wmma_f32_16x16x32_bf16(
          false, a1, false, b, (short)0, acc[1][t], false, false);
    }
    __syncthreads();  // all reads of this buffer done before TDM overwrites it
  }

  const int n = lane & 15, half = lane >> 4;
#pragma unroll
  for (int r = 0; r < 2; ++r)
#pragma unroll
    for (int t = 0; t < 4; ++t)
#pragma unroll
      for (int i = 0; i < 8; ++i) {
        const size_t off =
            (size_t)(bm + wm + r * 16 + half * 8 + i) * N + (bn + wn + t * 16 + n);
        if (OUT_BF16) ((bf16*)Cout)[off] = (bf16)acc[r][t][i];
        else          ((float*)Cout)[off] = acc[r][t][i];
      }
}

// ---------------------------------------------------------------------------
// Causal flash attention over bf16 qkv: [B, T, 3*D] (Q|K|V).  ctx out bf16
// [B,T,H,Hd].  128 threads = 4 waves, 64 q-rows/block.  Q/K tiles via TDM.
// ---------------------------------------------------------------------------
__global__ __launch_bounds__(128) void attn_flash(
    const bf16* __restrict__ qkv, bf16* __restrict__ ctx) {
  __shared__ __align__(16) bf16 sQ [64 * 64];     // [q][d]
  __shared__ __align__(16) bf16 sK [64 * 64];     // [k][d]
  __shared__ __align__(16) bf16 sVT[64 * 64];     // [d][k] (transposed)
  __shared__ __align__(16) bf16 sP [4][16 * 64];  // per-wave P staging [q][k]

  const int qt = blockIdx.x;
  const int h  = blockIdx.y;
  const int b  = blockIdx.z;
  const int tid = threadIdx.x, wave = tid >> 5, lane = tid & 31;
  const int n = lane & 15, half = lane >> 4;

  const size_t rowstride = 3 * D_MODEL;
  const bf16* Qg = qkv + (size_t)b * SEQ * rowstride + h * HEAD_DIM;
  const bf16* Kg = Qg + D_MODEL;
  const bf16* Vg = Qg + 2 * D_MODEL;

  // Q tile via TDM
  if (wave == 0) {
    tdm_load_2d(sQ, Qg + (size_t)(qt * 64) * rowstride,
                (unsigned)rowstride, 64, HEAD_DIM, 64, rowstride);
    __builtin_amdgcn_s_wait_tensorcnt(0);
  }
  __syncthreads();
  const v16bf qa0 = load_frag_k32(sQ + wave * 16 * 64, 64);
  const v16bf qa1 = load_frag_k32(sQ + wave * 16 * 64 + 32, 64);

  v8f o[4] = {};
  float mi[8], li[8];
#pragma unroll
  for (int i = 0; i < 8; ++i) { mi[i] = -1e30f; li[i] = 0.f; }
  const float scale = 0.125f;  // 1/sqrt(64)
  const int qrow0 = qt * 64 + wave * 16 + half * 8;

  for (int j = 0; j <= qt; ++j) {
    __syncthreads();  // prior reads of sK/sVT complete
    if (wave == 0)
      tdm_load_2d(sK, Kg + (size_t)(j * 64) * rowstride,
                  (unsigned)rowstride, 64, HEAD_DIM, 64, rowstride);
    // V tile, transposed into LDS (8 bf16 per 16B global load)
#pragma unroll
    for (int it = 0; it < 4; ++it) {
      const int id  = tid + it * 128;
      const int row = id >> 3;
      const int cg  = (id & 7) * 8;
      union { v4u u; bf16 e[8]; } vv;
      vv.u = *(const v4u*)(Vg + (size_t)(j * 64 + row) * rowstride + cg);
#pragma unroll
      for (int e = 0; e < 8; ++e) sVT[(cg + e) * 64 + row] = vv.e[e];
    }
    if (wave == 0) __builtin_amdgcn_s_wait_tensorcnt(0);
    __syncthreads();

    // S = Q K^T  (16 q-rows x 64 k-cols per wave)
    v8f s[4];
#pragma unroll
    for (int t = 0; t < 4; ++t) {
      v8f z = {};
      const v16bf b0 = load_frag_k32(sK + t * 16 * 64, 64);
      z = __builtin_amdgcn_wmma_f32_16x16x32_bf16(false, qa0, false, b0, (short)0, z, false, false);
      const v16bf b1 = load_frag_k32(sK + t * 16 * 64 + 32, 64);
      z = __builtin_amdgcn_wmma_f32_16x16x32_bf16(false, qa1, false, b1, (short)0, z, false, false);
      s[t] = z;
    }

    // Scale + causal mask
#pragma unroll
    for (int t = 0; t < 4; ++t) {
      const int kcol = j * 64 + t * 16 + n;
#pragma unroll
      for (int i = 0; i < 8; ++i) {
        const float v = s[t][i] * scale;
        s[t][i] = (kcol > qrow0 + i) ? -1e30f : v;
      }
    }

    // Online softmax (rows live in 16-lane groups -> xor masks 1,2,4,8)
#pragma unroll
    for (int i = 0; i < 8; ++i) {
      float rm = fmaxf(fmaxf(s[0][i], s[1][i]), fmaxf(s[2][i], s[3][i]));
      rm = fmaxf(rm, __shfl_xor(rm, 1, 32));
      rm = fmaxf(rm, __shfl_xor(rm, 2, 32));
      rm = fmaxf(rm, __shfl_xor(rm, 4, 32));
      rm = fmaxf(rm, __shfl_xor(rm, 8, 32));
      const float mnew = fmaxf(mi[i], rm);
      const float corr = __expf(mi[i] - mnew);
      float rs = 0.f;
#pragma unroll
      for (int t = 0; t < 4; ++t) {
        const float p = __expf(s[t][i] - mnew);
        s[t][i] = p;
        rs += p;
      }
      rs += __shfl_xor(rs, 1, 32);
      rs += __shfl_xor(rs, 2, 32);
      rs += __shfl_xor(rs, 4, 32);
      rs += __shfl_xor(rs, 8, 32);
      li[i] = li[i] * corr + rs;
      mi[i] = mnew;
#pragma unroll
      for (int t = 0; t < 4; ++t) o[t][i] *= corr;
    }

    // Stage P (C-layout regs -> row-major [q][k] bf16 in per-wave LDS)
#pragma unroll
    for (int t = 0; t < 4; ++t)
#pragma unroll
      for (int i = 0; i < 8; ++i)
        sP[wave][(half * 8 + i) * 64 + t * 16 + n] = (bf16)s[t][i];
    __syncthreads();

    // O += P V
    const v16bf pa0 = load_frag_k32(&sP[wave][0], 64);
    const v16bf pa1 = load_frag_k32(&sP[wave][0] + 32, 64);
#pragma unroll
    for (int t = 0; t < 4; ++t) {
      const v16bf vb0 = load_frag_k32(sVT + t * 16 * 64, 64);
      o[t] = __builtin_amdgcn_wmma_f32_16x16x32_bf16(false, pa0, false, vb0, (short)0, o[t], false, false);
      const v16bf vb1 = load_frag_k32(sVT + t * 16 * 64 + 32, 64);
      o[t] = __builtin_amdgcn_wmma_f32_16x16x32_bf16(false, pa1, false, vb1, (short)0, o[t], false, false);
    }
  }

  // Normalize, write ctx[B,T,H,Hd] as bf16
#pragma unroll
  for (int i = 0; i < 8; ++i) {
    const float inv = 1.0f / li[i];
    const int trow = qt * 64 + wave * 16 + half * 8 + i;
    bf16* dst = ctx + (size_t)(b * SEQ + trow) * D_MODEL + h * HEAD_DIM;
#pragma unroll
    for (int t = 0; t < 4; ++t)
      dst[t * 16 + n] = (bf16)(o[t][i] * inv);
  }
}

extern "C" void kernel_launch(void* const* d_in, const int* in_sizes, int n_in,
                              void* d_out, int out_size, void* d_ws, size_t ws_size,
                              hipStream_t stream) {
  (void)in_sizes; (void)n_in; (void)out_size; (void)ws_size;
  const float* x     = (const float*)d_in[0];
  const float* w_qkv = (const float*)d_in[1];
  const float* w_out = (const float*)d_in[2];
  float* out = (float*)d_out;

  const size_t n_x    = (size_t)BATCH * SEQ * D_MODEL;      // 4,194,304
  const size_t n_wqkv = (size_t)3 * D_MODEL * D_MODEL;      // 3,145,728
  const size_t n_wout = (size_t)D_MODEL * D_MODEL;          // 1,048,576
  const size_t n_qkv  = (size_t)BATCH * SEQ * 3 * D_MODEL;  // 12,582,912

  bf16* xb    = (bf16*)d_ws;
  bf16* wqkvb = xb + n_x;
  bf16* woutb = wqkvb + n_wqkv;
  bf16* qkvb  = woutb + n_wout;
  bf16* ctxb  = qkvb + n_qkv;

  // Stage 0: one-shot fp32 -> bf16 conversion of all operands
  cvt_f32_bf16<<<dim3(n_x    / 2048), 256, 0, stream>>>(x,     xb);
  cvt_f32_bf16<<<dim3(n_wqkv / 2048), 256, 0, stream>>>(w_qkv, wqkvb);
  cvt_f32_bf16<<<dim3(n_wout / 2048), 256, 0, stream>>>(w_out, woutb);

  // Stage 1: qkv = x @ w_qkv^T  (bf16 out)
  dim3 g1(BATCH * SEQ / 128, (3 * D_MODEL) / 128);
  gemm_bf16_tdm<true><<<g1, 256, 0, stream>>>(xb, wqkvb, qkvb,
                                              BATCH * SEQ, 3 * D_MODEL, D_MODEL);

  // Stage 2: causal flash attention
  dim3 g2(SEQ / 64, N_HEADS, BATCH);
  attn_flash<<<g2, 128, 0, stream>>>(qkvb, ctxb);

  // Stage 3: out = ctx @ w_out^T  (fp32 out)
  dim3 g3(BATCH * SEQ / 128, D_MODEL / 128);
  gemm_bf16_tdm<false><<<g3, 256, 0, stream>>>(ctxb, woutb, out,
                                               BATCH * SEQ, D_MODEL, D_MODEL);
}
